// HeteroGATEncoder_61151744360861
// MI455X (gfx1250) — matrix operand order repaired and checked by reading, not compiled, not run on hardware
//
#include <hip/hip_runtime.h>
#include <math.h>

// ---------------- static problem dims (match reference) ----------------
#define N_USERS 100000
#define N_ITEMS 20000
#define N_EDGES 800000
#define IN_DIM  256
#define HC      128   // HEADS*HID
#define HEADS   4
#define HID     32
#define OUT_DIM 32

typedef __attribute__((ext_vector_type(16))) __bf16 v16bf;
typedef __attribute__((ext_vector_type(8)))  float  v8f;

union BFrag { v16bf v; unsigned short s[16]; };

__device__ __forceinline__ unsigned short f2bf(float f) {
  union { float f; unsigned u; } c; c.f = f;
  unsigned u = c.u;
  u += 0x7fffu + ((u >> 16) & 1u);     // round-to-nearest-even
  return (unsigned short)(u >> 16);
}

// ---------------------------------------------------------------------
// WMMA bf16 GEMM: C[M,N] = A[M,K] @ W[K,N], A/W f32 row-major.
// W is transposed into LDS as bf16 [N][K+PAD] so each B fragment is two
// contiguous ds_load_b128; A fragment is 4x global_load_b128 (f32->bf16 in
// regs). 8 waves/block, 16 rows/wave, full 16xN tile in VGPRs.
//
// CDNA5 16-bit A 16x32 layout (ISA 7.12.2): half-lane g: halfs 0..7 hold
// K = g*8+0..7, halfs 8..15 hold K = 16+g*8+0..7 (contiguous pairs).
// A and B as loaded here: lane group g holds K = g*16 + half_index; A row / B col = lane%16.
// C/D: VGPR v -> M = v + 8*g, N = lane%16.
// ---------------------------------------------------------------------
#define LDS_PAD 16   // halfs; keeps 32B alignment, kills 512B-stride conflicts

template <int K, int N>
__global__ __launch_bounds__(256) void gemm_wmma_bf16(
    const float* __restrict__ A, const float* __restrict__ W,
    float* __restrict__ C, int M) {
  constexpr int NT = N / 16;
  constexpr int KP = K + LDS_PAD;
  extern __shared__ unsigned short ldsWT[];   // [N][KP] bf16, transposed

  const int tid = threadIdx.x;
  for (int i = tid; i < K * N; i += blockDim.x) {
    int k = i / N, n = i % N;                 // coalesced global read of W
    ldsWT[n * KP + k] = f2bf(W[i]);
  }
  __syncthreads();

  const int wave = tid >> 5;
  const int lane = tid & 31;
  const int g    = lane >> 4;
  const int mrow = lane & 15;
  const int row0 = (blockIdx.x * 8 + wave) * 16;
  if (row0 >= M) return;

  v8f acc[NT];
  #pragma unroll
  for (int t = 0; t < NT; ++t)
    #pragma unroll
    for (int j = 0; j < 8; ++j) acc[t][j] = 0.0f;

  int arow = row0 + mrow;
  if (arow >= M) arow = M - 1;                // clamp; stores are guarded
  const float* Arow = A + (size_t)arow * K;

  for (int kb = 0; kb < K; kb += 32) {
    // ---- A fragment: 4x float4, convert to bf16 ----
    const float4* Ak0 = (const float4*)(Arow + kb + g * 16);
    const float4* Ak1 = (const float4*)(Arow + kb + g * 16 + 8);
    float4 f0 = Ak0[0], f1 = Ak0[1], f2 = Ak1[0], f3 = Ak1[1];
    BFrag a;
    a.s[0]  = f2bf(f0.x); a.s[1]  = f2bf(f0.y); a.s[2]  = f2bf(f0.z); a.s[3]  = f2bf(f0.w);
    a.s[4]  = f2bf(f1.x); a.s[5]  = f2bf(f1.y); a.s[6]  = f2bf(f1.z); a.s[7]  = f2bf(f1.w);
    a.s[8]  = f2bf(f2.x); a.s[9]  = f2bf(f2.y); a.s[10] = f2bf(f2.z); a.s[11] = f2bf(f2.w);
    a.s[12] = f2bf(f3.x); a.s[13] = f2bf(f3.y); a.s[14] = f2bf(f3.z); a.s[15] = f2bf(f3.w);

    #pragma unroll
    for (int t = 0; t < NT; ++t) {
      // ---- B fragment: 16 contiguous halfs from transposed LDS ----
      const int ncol = t * 16 + mrow;
      v16bf bv = *(const v16bf*)&ldsWT[(size_t)ncol * KP + kb + g * 16];
      acc[t] = __builtin_amdgcn_wmma_f32_16x16x32_bf16(
          false, a.v, false, bv, (short)0, acc[t], false, false);
    }
  }

  if (row0 + 16 <= M) {
    // fast path: all 16 rows in range (every active wave here: M % 16 == 0) -> plain store clause,
    // row stride folds into immediate offsets
    #pragma unroll
    for (int t = 0; t < NT; ++t) {
      float* Cp = C + (size_t)(row0 + g * 8) * N + t * 16 + mrow;
      #pragma unroll
      for (int v = 0; v < 8; ++v) Cp[(size_t)v * N] = acc[t][v];
    }
  } else {
    #pragma unroll
    for (int t = 0; t < NT; ++t) {
      #pragma unroll
      for (int v = 0; v < 8; ++v) {
        int row = row0 + v + g * 8;
        if (row < M) C[(size_t)row * N + t * 16 + mrow] = acc[t][v];
      }
    }
  }
}

// ---------------------------------------------------------------------
__global__ void fill_f32(float* __restrict__ p, float v, int n) {
  int i = blockIdx.x * blockDim.x + threadIdx.x;
  if (i < n) p[i] = v;
}

// a[n,h] = sum_c h[n, h*32+c] * att[h*32+c]
__global__ void node_att_kernel(const float* __restrict__ h,
                                const float* __restrict__ att,
                                float* __restrict__ a, int Nn) {
  int t = blockIdx.x * blockDim.x + threadIdx.x;
  if (t >= Nn * HEADS) return;
  int n = t >> 2, hd = t & 3;
  const float4* p = (const float4*)(h + (size_t)n * HC + hd * HID);
  const float4* q = (const float4*)(att + hd * HID);
  float s = 0.f;
  #pragma unroll
  for (int i = 0; i < 8; ++i) {
    float4 x = p[i], y = q[i];
    s += x.x * y.x + x.y * y.y + x.z * y.z + x.w * y.w;
  }
  a[t] = s;
}

// Me[d][h] = sum_c W_edge[d, h*32+c] * att_edge[h*32+c]   (4x4 per conv)
__global__ void compute_me_kernel(const float* __restrict__ W_edge,
                                  const float* __restrict__ att_edge,
                                  float* __restrict__ Me) {
  int t = threadIdx.x;
  if (t >= 16) return;
  int d = t >> 2, hd = t & 3;
  float s = 0.f;
  for (int c = 0; c < HID; ++c)
    s += W_edge[d * HC + hd * HID + c] * att_edge[hd * HID + c];
  Me[d * HEADS + hd] = s;
}

__device__ __forceinline__ void atomicMaxF(float* addr, float val) {
  if (val >= 0.f) atomicMax((int*)addr, __float_as_int(val));
  else            atomicMin((unsigned int*)addr, __float_as_uint(val));
}

// pass 1: alpha = leaky_relu(a_src[s]+a_dst[d]+edge_attr@Me); seg-max via atomics
__global__ void edge_alpha_kernel(const float* __restrict__ edge_attr,
                                  const float* __restrict__ Me,
                                  const float* __restrict__ a_src,
                                  const float* __restrict__ a_dst,
                                  const int* __restrict__ sidx,
                                  const int* __restrict__ didx,
                                  float* __restrict__ alpha,
                                  float* __restrict__ amax) {
  int e = blockIdx.x * blockDim.x + threadIdx.x;
  if (e >= N_EDGES) return;
  float4 ea = ((const float4*)edge_attr)[e];
  int s = sidx[e], d = didx[e];
  #pragma unroll
  for (int h = 0; h < HEADS; ++h) {
    float ae = ea.x * Me[0 * HEADS + h] + ea.y * Me[1 * HEADS + h] +
               ea.z * Me[2 * HEADS + h] + ea.w * Me[3 * HEADS + h];
    float a = a_src[(size_t)s * HEADS + h] + a_dst[(size_t)d * HEADS + h] + ae;
    a = a > 0.f ? a : 0.2f * a;
    alpha[(size_t)e * HEADS + h] = a;
    atomicMaxF(&amax[(size_t)d * HEADS + h], a);
  }
}

// pass 2: ex = exp(alpha - amax[d]); seg-sum via atomics (ex overwrites alpha)
__global__ void edge_exp_kernel(const float* __restrict__ amax,
                                const int* __restrict__ didx,
                                float* __restrict__ alpha,
                                float* __restrict__ denom) {
  int e = blockIdx.x * blockDim.x + threadIdx.x;
  if (e >= N_EDGES) return;
  int d = didx[e];
  #pragma unroll
  for (int h = 0; h < HEADS; ++h) {
    float m = amax[(size_t)d * HEADS + h];
    m = (m > -1e30f) ? m : 0.f;
    float x = __expf(alpha[(size_t)e * HEADS + h] - m);
    alpha[(size_t)e * HEADS + h] = x;
    atomicAdd(&denom[(size_t)d * HEADS + h], x);
  }
}

// pass 3: out[d] += hs[s] * attn ; one wave32 per edge, 4 channels/lane
__global__ void edge_aggregate_kernel(const float* __restrict__ hs,
                                      const float* __restrict__ ex,
                                      const float* __restrict__ denom,
                                      const int* __restrict__ sidx,
                                      const int* __restrict__ didx,
                                      float* __restrict__ out) {
  int gid = blockIdx.x * blockDim.x + threadIdx.x;
  int e = gid >> 5, lane = gid & 31;
  if (e >= N_EDGES) return;
  int s = sidx[e], d = didx[e];
  int c = lane << 2;            // 4 contiguous channels, within one head
  int h = c >> 5;
  float attn = ex[(size_t)e * HEADS + h] /
               (denom[(size_t)d * HEADS + h] + 1e-16f);
  float4 v = *(const float4*)(hs + (size_t)s * HC + c);
  float* o = out + (size_t)d * HC + c;
  atomicAdd(o + 0, v.x * attn);
  atomicAdd(o + 1, v.y * attn);
  atomicAdd(o + 2, v.z * attn);
  atomicAdd(o + 3, v.w * attn);
}

__global__ void bias_elu_kernel(float* __restrict__ x,
                                const float* __restrict__ bias, int rows) {
  int i = blockIdx.x * blockDim.x + threadIdx.x;
  if (i >= rows * HC) return;
  float v = x[i] + bias[i & (HC - 1)];
  x[i] = v > 0.f ? v : (__expf(v) - 1.f);
}

// y = l2norm(x + b), rows of 32; one wave per row, lane-per-channel
__global__ void bias_l2norm_kernel(const float* __restrict__ x,
                                   const float* __restrict__ bias,
                                   float* __restrict__ out, int Nn) {
  int gid = blockIdx.x * blockDim.x + threadIdx.x;
  int n = gid >> 5, lane = gid & 31;
  if (n >= Nn) return;
  float v = x[(size_t)n * OUT_DIM + lane] + bias[lane];
  float ss = v * v;
  #pragma unroll
  for (int off = 16; off > 0; off >>= 1) ss += __shfl_xor(ss, off, 32);
  float norm = fmaxf(sqrtf(ss), 1e-12f);
  out[(size_t)n * OUT_DIM + lane] = v / norm;
}

// ---------------------------------------------------------------------
extern "C" void kernel_launch(void* const* d_in, const int* in_sizes, int n_in,
                              void* d_out, int out_size, void* d_ws, size_t ws_size,
                              hipStream_t stream) {
  (void)in_sizes; (void)n_in; (void)out_size; (void)ws_size;
  const float* x_user    = (const float*)d_in[0];
  const float* x_item    = (const float*)d_in[1];
  const float* edge_attr = (const float*)d_in[2];
  // 4 GAT param blocks: [W_src, W_dst, W_edge, att_src, att_dst, att_edge, bias]
  const float* P[4][7];
  int idx = 3;
  for (int l = 0; l < 4; ++l)
    for (int j = 0; j < 7; ++j) P[l][j] = (const float*)d_in[idx++];
  const float* upW = (const float*)d_in[idx++];
  const float* upB = (const float*)d_in[idx++];
  const float* ipW = (const float*)d_in[idx++];
  const float* ipB = (const float*)d_in[idx++];
  const int* edge_src = (const int*)d_in[idx++];
  const int* edge_dst = (const int*)d_in[idx++];

  // ---- workspace bump allocation (floats) ----
  float* w = (float*)d_ws;
  auto alloc = [&](size_t n) { float* p = w; w += n; return p; };
  float* bufA   = alloc((size_t)N_USERS * HC);  // hs (user source)
  float* bufB   = alloc((size_t)N_ITEMS * HC);  // hd (item dest)
  float* bufC   = alloc((size_t)N_ITEMS * HC);  // hs (item source)
  float* bufD   = alloc((size_t)N_USERS * HC);  // hd (user dest)
  float* user_h = alloc((size_t)N_USERS * HC);
  float* item_h = alloc((size_t)N_ITEMS * HC);
  float* alpha  = alloc((size_t)N_EDGES * HEADS);
  float* a_src  = alloc((size_t)N_USERS * HEADS);
  float* a_dst  = alloc((size_t)N_USERS * HEADS);
  float* amax   = alloc((size_t)N_USERS * HEADS);
  float* denom  = alloc((size_t)N_USERS * HEADS);
  float* Me     = alloc(16);

  auto gemm = [&](const float* A, const float* W_, float* C, int M, int K, int N) {
    dim3 grid((M + 127) / 128);
    size_t shm;
    if (K == 256 && N == 128) {
      shm = (size_t)128 * (256 + LDS_PAD) * 2;
      gemm_wmma_bf16<256, 128><<<grid, 256, shm, stream>>>(A, W_, C, M);
    } else if (K == 128 && N == 128) {
      shm = (size_t)128 * (128 + LDS_PAD) * 2;
      gemm_wmma_bf16<128, 128><<<grid, 256, shm, stream>>>(A, W_, C, M);
    } else {  // K == 128, N == 32
      shm = (size_t)32 * (128 + LDS_PAD) * 2;
      gemm_wmma_bf16<128, 32><<<grid, 256, shm, stream>>>(A, W_, C, M);
    }
  };
  auto fill = [&](float* p, float v, int n) {
    fill_f32<<<(n + 255) / 256, 256, 0, stream>>>(p, v, n);
  };
  auto natt = [&](const float* h, const float* att, float* a, int Nn) {
    node_att_kernel<<<(Nn * HEADS + 255) / 256, 256, 0, stream>>>(h, att, a, Nn);
  };
  const int EB = (N_EDGES + 255) / 256;
  const int EWB = (N_EDGES * 32 + 255) / 256;
  auto edge_phase = [&](const float* hs, const float* W_edge, const float* att_edge,
                        const int* si, const int* di, int Nd, float* out) {
    compute_me_kernel<<<1, 16, 0, stream>>>(W_edge, att_edge, Me);
    fill(amax, -INFINITY, Nd * HEADS);
    fill(denom, 0.f, Nd * HEADS);
    fill(out, 0.f, Nd * HC);
    edge_alpha_kernel<<<EB, 256, 0, stream>>>(edge_attr, Me, a_src, a_dst, si, di, alpha, amax);
    edge_exp_kernel<<<EB, 256, 0, stream>>>(amax, di, alpha, denom);
    edge_aggregate_kernel<<<EWB, 256, 0, stream>>>(hs, alpha, denom, si, di, out);
  };

  // ======== layer 0 ========
  gemm(x_user, P[0][0], bufA, N_USERS, IN_DIM, HC);
  gemm(x_item, P[0][1], bufB, N_ITEMS, IN_DIM, HC);
  gemm(x_item, P[1][0], bufC, N_ITEMS, IN_DIM, HC);
  gemm(x_user, P[1][1], bufD, N_USERS, IN_DIM, HC);
  // l0_rates: user -> item
  natt(bufA, P[0][3], a_src, N_USERS);
  natt(bufB, P[0][4], a_dst, N_ITEMS);
  edge_phase(bufA, P[0][2], P[0][5], edge_src, edge_dst, N_ITEMS, item_h);
  // l0_rev: item -> user (flipped edges)
  natt(bufC, P[1][3], a_src, N_ITEMS);
  natt(bufD, P[1][4], a_dst, N_USERS);
  edge_phase(bufC, P[1][2], P[1][5], edge_dst, edge_src, N_USERS, user_h);
  bias_elu_kernel<<<(N_ITEMS * HC + 255) / 256, 256, 0, stream>>>(item_h, P[0][6], N_ITEMS);
  bias_elu_kernel<<<(N_USERS * HC + 255) / 256, 256, 0, stream>>>(user_h, P[1][6], N_USERS);

  // ======== layer 1 (all 4 GEMMs first: user_h/item_h are inputs, then reused as outputs) ====
  gemm(user_h, P[2][0], bufA, N_USERS, HC, HC);
  gemm(item_h, P[2][1], bufB, N_ITEMS, HC, HC);
  gemm(item_h, P[3][0], bufC, N_ITEMS, HC, HC);
  gemm(user_h, P[3][1], bufD, N_USERS, HC, HC);
  // l1_rates -> item_h
  natt(bufA, P[2][3], a_src, N_USERS);
  natt(bufB, P[2][4], a_dst, N_ITEMS);
  edge_phase(bufA, P[2][2], P[2][5], edge_src, edge_dst, N_ITEMS, item_h);
  // l1_rev -> user_h
  natt(bufC, P[3][3], a_src, N_ITEMS);
  natt(bufD, P[3][4], a_dst, N_USERS);
  edge_phase(bufC, P[3][2], P[3][5], edge_dst, edge_src, N_USERS, user_h);
  bias_elu_kernel<<<(N_ITEMS * HC + 255) / 256, 256, 0, stream>>>(item_h, P[2][6], N_ITEMS);
  bias_elu_kernel<<<(N_USERS * HC + 255) / 256, 256, 0, stream>>>(user_h, P[3][6], N_USERS);

  // ======== projections + l2norm ========
  float* utmp = bufA;   // reuse
  float* itmp = bufB;
  gemm(user_h, upW, utmp, N_USERS, HC, OUT_DIM);
  gemm(item_h, ipW, itmp, N_ITEMS, HC, OUT_DIM);
  float* out = (float*)d_out;
  bias_l2norm_kernel<<<(N_USERS * 32 + 255) / 256, 256, 0, stream>>>(utmp, upB, out, N_USERS);
  bias_l2norm_kernel<<<(N_ITEMS * 32 + 255) / 256, 256, 0, stream>>>(
      itmp, ipB, out + (size_t)N_USERS * OUT_DIM, N_ITEMS);
}
